// CustomGATLayerEdgeReprFeat_25632364822808
// MI455X (gfx1250) — matrix operand order repaired and checked by reading, not compiled, not run on hardware
//
#include <hip/hip_runtime.h>
#include <hip/hip_bf16.h>

// ---------------------------------------------------------------------------
// GAT layer w/ edge features for MI455X (gfx1250, wave32, WMMA).
// N=50000, E=600000, IN=128, O=16, H=8  (IN = H*O = 128, both N,E % 16 == 0)
//
// Weight matrices are pre-swizzled into WMMA B-fragment layout so each lane
// loads its 16 bf16 B elements as one contiguous 32-byte access.
// ---------------------------------------------------------------------------

typedef __attribute__((ext_vector_type(16))) __bf16 v16bf;
typedef __attribute__((ext_vector_type(8)))  float  v8f;

__device__ __forceinline__ unsigned enc_ord(float x) {
  // monotonic float->uint map so float-max == uint atomicMax
  unsigned u = __float_as_uint(x);
  return (u & 0x80000000u) ? ~u : (u | 0x80000000u);
}
__device__ __forceinline__ float dec_ord(unsigned u) {
  return (u & 0x80000000u) ? __uint_as_float(u & 0x7FFFFFFFu)
                           : __uint_as_float(~u);
}
#define ENC_NEG_INF 0x007FFFFFu  // enc_ord(-inf)

__device__ __forceinline__ float elu1(float x) {
  return x > 0.f ? x : (__expf(x) - 1.f);
}

// ---------------------------------------------------------------------------
// k_prep: repack weights to bf16 in *pre-swizzled WMMA B-fragment* layout.
//
// B-fragment layout (ISA 7.12.2, wave32, 16-bit B, 32x16 KxN):
//   lane l: n = l&15, kh = l>>4; element j <-> K = j<8 ? kh*8+j : 16+kh*8+(j-8).
// Packed buffer: Bpk[frag][lane][j]  (16 bf16 = 32 contiguous bytes per lane)
//
//   WhPk/WePk: frag = nt*4 + kb  (nt = 16-col tile 0..7, kb = K-block 0..3)
//              value = W[h=nt][i = kb*32 + (j<8 ? kh*8+j : 16+kh*8+j-8)][o = n]
//   WpPk:      frag = h*2 + half; K' = half*32 + (j<8 ? kh*8+j : 16+kh*8+j-8)
//              value = K' < 48 ? Wproj[h][K'][n] : 0     (K 48..63 zero-pad)
// ---------------------------------------------------------------------------
__global__ void k_prep(const float* __restrict__ Wh, const float* __restrict__ We,
                       const float* __restrict__ Wproj,
                       __bf16* __restrict__ WhPk, __bf16* __restrict__ WePk,
                       __bf16* __restrict__ WpPk) {
  int i = blockIdx.x * 256 + threadIdx.x;
  if (i < 4 * 8 * 32 * 16) {                 // 16384 elements
    int j    = i & 15;
    int lane = (i >> 4) & 31;
    int kb   = (i >> 9) & 3;
    int nt   = i >> 11;
    int n = lane & 15, kh = (lane >> 4) & 1;
    int K = kb * 32 + (j < 8 ? kh * 8 + j : 16 + kh * 8 + (j - 8));
    // Wh/We: [H][128][16] row-major; col of flattened B = nt*16 + n -> h=nt,o=n
    WhPk[i] = (__bf16)Wh[nt * 2048 + K * 16 + n];
    WePk[i] = (__bf16)We[nt * 2048 + K * 16 + n];
  }
  if (i < 8 * 2 * 32 * 16) {                 // 8192 elements
    int j    = i & 15;
    int lane = (i >> 4) & 31;
    int half = (i >> 9) & 1;
    int hq   = i >> 10;
    int n = lane & 15, kh = (lane >> 4) & 1;
    int K = half * 32 + (j < 8 ? kh * 8 + j : 16 + kh * 8 + (j - 8));
    WpPk[i] = (__bf16)(K < 48 ? Wproj[hq * 768 + K * 16 + n] : 0.f);
  }
}

// load one pre-swizzled B fragment: 32 contiguous bytes per lane
__device__ __forceinline__ v16bf load_B_pk(const __bf16* __restrict__ Bpk,
                                           int frag, int lane) {
  return *(const v16bf*)(Bpk + (size_t)(frag * 32 + lane) * 16);
}

// ---------------------------------------------------------------------------
// k_init: zero accumulators, seed segment-max with encoded -inf.
// ---------------------------------------------------------------------------
__global__ void k_init(unsigned* __restrict__ mbuf, float* __restrict__ denom,
                       float* __restrict__ hacc, int nm, int nh) {
  int i = blockIdx.x * 256 + threadIdx.x;
  if (i < nh) hacc[i] = 0.f;
  if (i < nm) { mbuf[i] = ENC_NEG_INF; denom[i] = 0.f; }
}

// ---------------------------------------------------------------------------
// k_node_gemm: z_h[N,128] = h[N,128] @ Wh_flat[128,128]  (bf16 WMMA, f32 acc)
// one wave per 16-row tile; 8 col tiles x 4 K-blocks = 32 WMMAs/wave.
// ---------------------------------------------------------------------------
__global__ __launch_bounds__(128) void k_node_gemm(
    const float* __restrict__ h, const __bf16* __restrict__ WhPk,
    float* __restrict__ zh, int ntiles) {
  int wave = (blockIdx.x * blockDim.x + threadIdx.x) >> 5;
  int lane = threadIdx.x & 31;
  if (wave >= ntiles) return;
  int row0 = wave * 16;
  int r = lane & 15, kh = lane >> 4;

  // A fragments (16x32 bf16): lane covers row r; elems 0..7 -> K=kb*32+kh*8+i,
  // elems 8..15 -> K=kb*32+16+kh*8+i  (ISA 7.12.2 16-bit A layout).
  v16bf a[4];
  const float* hp = h + (size_t)(row0 + r) * 128;
#pragma unroll
  for (int kb = 0; kb < 4; ++kb) {
    const float* p0 = hp + kb * 32 + kh * 8;
#pragma unroll
    for (int i = 0; i < 8; ++i) {
      a[kb][i]     = (__bf16)p0[i];
      a[kb][8 + i] = (__bf16)p0[16 + i];
    }
  }
#pragma unroll
  for (int nt = 0; nt < 8; ++nt) {
    v8f acc = {};
#pragma unroll
    for (int kb = 0; kb < 4; ++kb) {
      v16bf b = load_B_pk(WhPk, nt * 4 + kb, lane);
      acc = __builtin_amdgcn_wmma_f32_16x16x32_bf16(false, a[kb], false, b,
                                                    (short)0, acc, false, false);
    }
    // C layout: vgpr v -> row v + 8*kh, col = r
    float* zp = zh + (size_t)(row0 + 8 * kh) * 128 + nt * 16 + r;
#pragma unroll
    for (int v = 0; v < 8; ++v) zp[v * 128] = acc[v];
  }
}

// ---------------------------------------------------------------------------
// k_edge1: z_e = e @ We (WMMA, store bf16), attention logits + atomicMax.
// ---------------------------------------------------------------------------
__global__ __launch_bounds__(256) void k_edge1(
    const float* __restrict__ e, const int* __restrict__ src,
    const int* __restrict__ dst, const __bf16* __restrict__ WePk,
    const float* __restrict__ zh, const float* __restrict__ Wattn,
    __bf16* __restrict__ zeb, float* __restrict__ attn,
    unsigned* __restrict__ mbuf, int ntiles) {
  int wave = (blockIdx.x * blockDim.x + threadIdx.x) >> 5;
  int lane = threadIdx.x & 31;
  if (wave >= ntiles) return;
  size_t row0 = (size_t)wave * 16;
  int r = lane & 15, kh = lane >> 4;

  // prefetch next tile's e rows into cache hierarchy
  if (wave + 1 < ntiles)
    __builtin_prefetch(e + (row0 + 16 + r) * 128 + kh * 64, 0, 0);

  v16bf a[4];
  const float* ep = e + (row0 + r) * 128;
#pragma unroll
  for (int kb = 0; kb < 4; ++kb) {
    const float* p0 = ep + kb * 32 + kh * 8;
#pragma unroll
    for (int i = 0; i < 8; ++i) {
      a[kb][i]     = (__bf16)p0[i];
      a[kb][8 + i] = (__bf16)p0[16 + i];
    }
  }
#pragma unroll
  for (int nt = 0; nt < 8; ++nt) {
    v8f acc = {};
#pragma unroll
    for (int kb = 0; kb < 4; ++kb) {
      v16bf b = load_B_pk(WePk, nt * 4 + kb, lane);
      acc = __builtin_amdgcn_wmma_f32_16x16x32_bf16(false, a[kb], false, b,
                                                    (short)0, acc, false, false);
    }
    __bf16* zp = zeb + (row0 + 8 * kh) * 128 + nt * 16 + r;
#pragma unroll
    for (int v = 0; v < 8; ++v) zp[v * 128] = (__bf16)acc[v];
  }
  // make the just-written z_e tile visible to this wave's reloads
  asm volatile("s_wait_storecnt 0" ::: "memory");

  // attention: lane handles edge r for heads kh*4 .. kh*4+3
  size_t edge = row0 + r;
  int s = src[edge], d = dst[edge];
  const __bf16* zrow = zeb + edge * 128;
  const float* zs = zh + (size_t)s * 128;
  const float* zd = zh + (size_t)d * 128;
#pragma unroll
  for (int hh = 0; hh < 4; ++hh) {
    int hq = kh * 4 + hh;
    const float* wa = Wattn + hq * 48;
    v16bf zv = *(const v16bf*)(zrow + hq * 16);   // 32B contiguous
    float sum = 0.f;
#pragma unroll
    for (int o = 0; o < 16; ++o) {
      sum += (float)zv[o] * wa[o];
      sum += zs[hq * 16 + o] * wa[16 + o];
      sum += zd[hq * 16 + o] * wa[32 + o];
    }
    float av = sum > 0.f ? sum : 0.01f * sum;  // leaky_relu(0.01)
    attn[edge * 8 + hq] = av;
    atomicMax(&mbuf[(size_t)d * 8 + hq], enc_ord(av));
  }
}

// ---------------------------------------------------------------------------
// k_edge2: e_proj via WMMA (+bias, elu, residual -> e_out); exp/denom;
//          unnormalized aggregation hacc += ex * z_src  (scatter atomics).
// ---------------------------------------------------------------------------
__global__ __launch_bounds__(256) void k_edge2(
    const float* __restrict__ e, const int* __restrict__ src,
    const int* __restrict__ dst, const float* __restrict__ zh,
    const __bf16* __restrict__ zeb, const float* __restrict__ attn,
    const unsigned* __restrict__ mbuf, const __bf16* __restrict__ WpPk,
    const float* __restrict__ bproj, float* __restrict__ denom,
    float* __restrict__ hacc, float* __restrict__ e_out, int ntiles) {
  __shared__ float sEx[8][16][8];
  __shared__ int sSrc[8][16];
  __shared__ int sDst[8][16];

  int w = threadIdx.x >> 5, lane = threadIdx.x & 31;
  int wave = blockIdx.x * 8 + w;
  bool active = wave < ntiles;
  int r = lane & 15, kh = lane >> 4;
  size_t row0 = (size_t)wave * 16;
  size_t edge = row0 + r;

  if (active) {
    int s = src[edge], d = dst[edge];
    if (kh == 0) { sSrc[w][r] = s; sDst[w][r] = d; }

    // softmax numerator + denominator accumulation
#pragma unroll
    for (int hh = 0; hh < 4; ++hh) {
      int hq = kh * 4 + hh;
      float av = attn[edge * 8 + hq];
      float mf = dec_ord(mbuf[(size_t)d * 8 + hq]);
      if ((__float_as_uint(mf) & 0x7F800000u) == 0x7F800000u) mf = 0.f;  // guard
      float ex = __expf(av - mf);
      atomicAdd(&denom[(size_t)d * 8 + hq], ex);
      sEx[w][r][hq] = ex;
    }

    // e_proj per head: [z_e|z_src] @ Wproj[0:32] + [z_dst|0] @ Wproj[32:64]
    const __bf16* ze_row = zeb + edge * 128;
    const float* zs_row = zh + (size_t)s * 128;
    const float* zd_row = zh + (size_t)d * 128;
#pragma unroll
    for (int hq = 0; hq < 8; ++hq) {
      v16bf a1, a2;
#pragma unroll
      for (int i = 0; i < 8; ++i) {
        a1[i]     = ze_row[hq * 16 + kh * 8 + i];            // K 0..15  (z_e)
        a1[8 + i] = (__bf16)zs_row[hq * 16 + kh * 8 + i];    // K 16..31 (z_src)
        a2[i]     = (__bf16)zd_row[hq * 16 + kh * 8 + i];    // K 0..15  (z_dst)
        a2[8 + i] = (__bf16)0.f;                             // K 16..31 (x zero B)
      }
      v16bf b1 = load_B_pk(WpPk, hq * 2 + 0, lane);   // Wproj rows 0..31
      v16bf b2 = load_B_pk(WpPk, hq * 2 + 1, lane);   // Wproj rows 32..63
      float bias = bproj[hq * 16 + r];
      v8f acc;
#pragma unroll
      for (int v = 0; v < 8; ++v) acc[v] = bias;
      acc = __builtin_amdgcn_wmma_f32_16x16x32_bf16(false, a1, false, b1,
                                                    (short)0, acc, false, false);
      acc = __builtin_amdgcn_wmma_f32_16x16x32_bf16(false, a2, false, b2,
                                                    (short)0, acc, false, false);
      const float* erow = e + (row0 + 8 * kh) * 128 + hq * 16 + r;
      float* orow = e_out + (row0 + 8 * kh) * 128 + hq * 16 + r;
#pragma unroll
      for (int v = 0; v < 8; ++v)
        orow[v * 128] = erow[v * 128] + elu1(acc[v]);
    }
  }

  __syncthreads();

  if (active) {
    // 16 edges x 128 cols = 2048 scatter-adds per wave, 64 per lane
#pragma unroll 4
    for (int it = 0; it < 64; ++it) {
      int flat = it * 32 + lane;
      int ei = flat >> 7, c = flat & 127;
      int hq = c >> 4;
      float ex = sEx[w][ei][hq];
      int sn = sSrc[w][ei], dn = sDst[w][ei];
      atomicAdd(&hacc[(size_t)dn * 128 + c], ex * zh[(size_t)sn * 128 + c]);
    }
  }
}

// ---------------------------------------------------------------------------
// k_node_out: h_out = h + elu(hacc / max(denom, 1e-9))
// ---------------------------------------------------------------------------
__global__ void k_node_out(const float* __restrict__ h,
                           const float* __restrict__ hacc,
                           const float* __restrict__ denom,
                           float* __restrict__ out, int total) {
  int i = blockIdx.x * 256 + threadIdx.x;
  if (i >= total) return;
  int n = i >> 7, hq = (i >> 4) & 7;
  float den = denom[n * 8 + hq];
  float ag = hacc[i] / fmaxf(den, 1e-9f);
  out[i] = h[i] + elu1(ag);
}

// ---------------------------------------------------------------------------
extern "C" void kernel_launch(void* const* d_in, const int* in_sizes, int n_in,
                              void* d_out, int out_size, void* d_ws, size_t ws_size,
                              hipStream_t stream) {
  const float* h     = (const float*)d_in[0];
  const float* e     = (const float*)d_in[1];
  const int*   src   = (const int*)d_in[2];
  const int*   dst   = (const int*)d_in[3];
  const float* Wh    = (const float*)d_in[4];
  const float* We    = (const float*)d_in[5];
  const float* Wproj = (const float*)d_in[6];
  const float* bproj = (const float*)d_in[7];
  const float* Wattn = (const float*)d_in[8];

  const int N = in_sizes[0] / 128;
  const int E = in_sizes[1] / 128;
  const int nodeTiles = N / 16;   // N % 16 == 0
  const int edgeTiles = E / 16;   // E % 16 == 0

  float* h_out = (float*)d_out;
  float* e_out = (float*)d_out + (size_t)N * 128;

  char* ws = (char*)d_ws;
  float*    zh    = (float*)ws;    ws += (size_t)N * 128 * 4;
  __bf16*   zeb   = (__bf16*)ws;   ws += (size_t)E * 128 * 2;
  float*    attn  = (float*)ws;    ws += (size_t)E * 8 * 4;
  unsigned* mbuf  = (unsigned*)ws; ws += (size_t)N * 8 * 4;
  float*    denom = (float*)ws;    ws += (size_t)N * 8 * 4;
  float*    hacc  = (float*)ws;    ws += (size_t)N * 128 * 4;
  __bf16*   WhPk  = (__bf16*)ws;   ws += 4 * 8 * 32 * 16 * 2;
  __bf16*   WePk  = (__bf16*)ws;   ws += 4 * 8 * 32 * 16 * 2;
  __bf16*   WpPk  = (__bf16*)ws;   ws += 8 * 2 * 32 * 16 * 2;

  k_prep<<<64, 256, 0, stream>>>(Wh, We, Wproj, WhPk, WePk, WpPk);
  k_init<<<(N * 128 + 255) / 256, 256, 0, stream>>>(mbuf, denom, hacc,
                                                    N * 8, N * 128);
  k_node_gemm<<<(nodeTiles + 3) / 4, 128, 0, stream>>>(h, WhPk, zh, nodeTiles);
  k_edge1<<<(edgeTiles + 7) / 8, 256, 0, stream>>>(e, src, dst, WePk, zh, Wattn,
                                                   zeb, attn, mbuf, edgeTiles);
  k_edge2<<<(edgeTiles + 7) / 8, 256, 0, stream>>>(e, src, dst, zh, zeb, attn,
                                                   mbuf, WpPk, bproj, denom,
                                                   hacc, e_out, edgeTiles);
  k_node_out<<<(N * 128 + 255) / 256, 256, 0, stream>>>(h, hacc, denom, h_out,
                                                        N * 128);
}
